// AttentionWithFastKANTransform_53996328845912
// MI455X (gfx1250) — matrix operand order repaired and checked by reading, not compiled, not run on hardware
//
#include <hip/hip_runtime.h>
#include <hip/hip_bf16.h>
#include <stdint.h>

typedef __attribute__((ext_vector_type(16))) __bf16 v16bf;
typedef __attribute__((ext_vector_type(8)))  float  v8f;
typedef __attribute__((ext_vector_type(8)))  unsigned short u16x8;

#define DEV static __device__ __forceinline__

DEV unsigned short f32_bf16(float f){
  unsigned int u = __float_as_uint(f);
  u += 0x7FFFu + ((u >> 16) & 1u);           // round-to-nearest-even
  return (unsigned short)(u >> 16);
}

union BF16Pack { u16x8 h[2]; v16bf v; };

// Build a 16-element bf16 A/B fragment from two 16-byte chunks.
DEV v16bf ld_bf16x16(const unsigned short* p0, const unsigned short* p1){
  BF16Pack u;
  u.h[0] = *(const u16x8*)p0;
  u.h[1] = *(const u16x8*)p1;
  return u.v;
}

DEV v8f wmma_bf16(v16bf a, v16bf b, v8f c){
  return __builtin_amdgcn_wmma_f32_16x16x32_bf16(false, a, false, b, (short)0, c, false, false);
}

// ---------------------------------------------------------------------------
// FastKAN: Y[tok][o] = sum_e basis[tok][e]*spline[o][e] + sum_i silu(xn)[i]*baseT[o][i] + b[o]
// 16-token tile per block, 8 waves, each wave owns 32 output columns.
// ---------------------------------------------------------------------------
__global__ __launch_bounds__(256)
void fastkan_kernel(const float* __restrict__ X,
                    const float* __restrict__ ln_s, const float* __restrict__ ln_b,
                    const float* __restrict__ gridv, const float* __restrict__ denom,
                    const unsigned short* __restrict__ splineW,   // [256][2048] bf16
                    const unsigned short* __restrict__ baseT,     // [256][256]  bf16 (out-major)
                    const float* __restrict__ base_b,
                    float outScale,
                    unsigned short* __restrict__ out_bf16,
                    float* __restrict__ out_f32)
{
  __shared__ __align__(16) float sX[16*256];            // raw x, then xn
  __shared__ __align__(16) unsigned short sBasis[16*1024]; // half of the 16x2048 basis
  __shared__ __align__(16) unsigned short sSilu[16*256];
  __shared__ float sRed[512];
  __shared__ float sMu[16], sRstd[16];

  const int tid = threadIdx.x;
  const int row = tid >> 4, sub = tid & 15;
  const size_t tokbase = (size_t)blockIdx.x * 16;

  { // load 16x256 tile
    const float* src = X + (tokbase + row) * 256 + sub * 16;
    float* dst = &sX[row*256 + sub*16];
    #pragma unroll
    for (int i=0;i<16;++i) dst[i] = src[i];
  }
  __syncthreads();
  { // layernorm partials
    float s=0.f, ss=0.f;
    const float* p = &sX[row*256 + sub*16];
    #pragma unroll
    for (int i=0;i<16;++i){ float vv=p[i]; s+=vv; ss+=vv*vv; }
    sRed[row*16+sub]=s; sRed[256+row*16+sub]=ss;
  }
  __syncthreads();
  if (tid < 16){
    float S=0.f, SS=0.f;
    #pragma unroll
    for (int j=0;j<16;++j){ S+=sRed[tid*16+j]; SS+=sRed[256+tid*16+j]; }
    float mu  = S * (1.0f/256.0f);
    float var = SS * (1.0f/256.0f) - mu*mu;
    sMu[tid]=mu; sRstd[tid]=rsqrtf(var + 1e-6f);
  }
  __syncthreads();
  { // xn (in place) + silu(xn) as bf16
    float mu=sMu[row], rstd=sRstd[row];
    #pragma unroll
    for (int i=0;i<16;++i){
      int c = sub*16+i;
      float x  = sX[row*256+c];
      float xn = (x-mu)*rstd*ln_s[c] + ln_b[c];
      sX[row*256+c]=xn;
      float sig = 1.0f/(1.0f + __expf(-xn));
      sSilu[row*256+c] = f32_bf16(xn*sig);
    }
  }
  __syncthreads();

  const int wave = tid>>5, lane = tid&31;
  const int lr = lane&15, hi = lane>>4;
  const int col0 = wave*32;
  v8f acc0 = {}, acc1 = {};
  float gr[8];
  #pragma unroll
  for (int g=0; g<8; ++g) gr[g] = gridv[g];
  const float dinv = 1.0f/denom[0];

  for (int phase=0; phase<2; ++phase){
    { // fill basis half: expanded index e in [phase*1024, phase*1024+1024)
      int c0 = phase*128 + sub*8;
      #pragma unroll
      for (int i=0;i<8;++i){
        int c = c0+i;
        float xn = sX[row*256+c];
        int eb = (c - phase*128) << 3;
        #pragma unroll
        for (int g=0; g<8; ++g){
          float t = (xn - gr[g])*dinv;
          sBasis[row*1024 + eb + g] = f32_bf16(__expf(-t*t));
        }
      }
    }
    __syncthreads();
    const unsigned short* arow = &sBasis[lr*1024];
    for (int kt=0; kt<1024; kt+=32){
      // A-layout: lanes 0-15 hold K {0..15}, lanes 16-31 hold K {16..31}
      v16bf A = ld_bf16x16(arow + kt + hi*16, arow + kt + hi*16 + 8);
      size_t kg = (size_t)phase*1024 + kt + (size_t)hi*16;
      const unsigned short* b0 = splineW + (size_t)(col0+lr)*2048 + kg;
      const unsigned short* b1 = splineW + (size_t)(col0+16+lr)*2048 + kg;
      acc0 = wmma_bf16(A, ld_bf16x16(b0, b0+8), acc0);
      acc1 = wmma_bf16(A, ld_bf16x16(b1, b1+8), acc1);
    }
    __syncthreads();
  }
  { // base matmul, K = 256
    const unsigned short* arow = &sSilu[lr*256];
    for (int kt=0; kt<256; kt+=32){
      v16bf A = ld_bf16x16(arow + kt + hi*16, arow + kt + hi*16 + 8);
      const unsigned short* b0 = baseT + (size_t)(col0+lr)*256 + kt + hi*16;
      const unsigned short* b1 = baseT + (size_t)(col0+16+lr)*256 + kt + hi*16;
      acc0 = wmma_bf16(A, ld_bf16x16(b0, b0+8), acc0);
      acc1 = wmma_bf16(A, ld_bf16x16(b1, b1+8), acc1);
    }
  }
  float bb0 = base_b[col0+lr], bb1 = base_b[col0+16+lr];
  #pragma unroll
  for (int j=0;j<8;++j){
    int rm = j + 8*hi;                 // C layout: VGPR j -> row j (+8 for upper lanes)
    size_t o = (tokbase + rm)*256;
    float r0 = (acc0[j] + bb0) * outScale;
    float r1 = (acc1[j] + bb1) * outScale;
    if (out_bf16){ out_bf16[o + col0 + lr] = f32_bf16(r0); out_bf16[o + col0 + 16 + lr] = f32_bf16(r1); }
    if (out_f32 ){ out_f32 [o + col0 + lr] = r0;           out_f32 [o + col0 + 16 + lr] = r1; }
  }
}

// ---------------------------------------------------------------------------
// Flash attention: 1 wave per (b, h, 16-query tile); online softmax over keys.
// ---------------------------------------------------------------------------
__global__ __launch_bounds__(32)
void attn_kernel(const unsigned short* __restrict__ wq,
                 const unsigned short* __restrict__ wk,
                 const unsigned short* __restrict__ wvT,   // [b][h][d(32)][key(2048)]
                 float* __restrict__ out)                  // [b][q][256] f32
{
  const int L = 2048;
  const int lane = threadIdx.x;
  const int lr = lane & 15, hi = lane >> 4;
  const int qb = blockIdx.x * 16;
  const int h  = blockIdx.y;
  const int b  = blockIdx.z;

  __shared__ __align__(16) unsigned short sP[16*32];

  const unsigned short* qrow = wq + ((size_t)(b*L + qb + lr))*256 + h*32 + hi*16;
  v16bf Aq = ld_bf16x16(qrow, qrow + 8);
  v8f o0 = {}, o1 = {};
  float m8[8], l8[8];
  #pragma unroll
  for (int j=0;j<8;++j){ m8[j] = -3.0e38f; l8[j] = 0.0f; }

  const unsigned short* vbase = wvT + ((size_t)(b*8 + h) * 32) * L;

  for (int kb = 0; kb < L; kb += 32){
    const unsigned short* k0p = wk + ((size_t)(b*L + kb + lr     ))*256 + h*32 + hi*16;
    const unsigned short* k1p = wk + ((size_t)(b*L + kb + 16 + lr))*256 + h*32 + hi*16;
    v8f z = {};
    v8f s0 = wmma_bf16(Aq, ld_bf16x16(k0p, k0p+8), z);   // 16q x 16k, K=HD=32
    v8f s1 = wmma_bf16(Aq, ld_bf16x16(k1p, k1p+8), z);

    #pragma unroll
    for (int j=0;j<8;++j){
      float a = s0[j], c = s1[j];
      float mx = fmaxf(a, c);
      mx = fmaxf(mx, __shfl_xor(mx, 1));
      mx = fmaxf(mx, __shfl_xor(mx, 2));
      mx = fmaxf(mx, __shfl_xor(mx, 4));
      mx = fmaxf(mx, __shfl_xor(mx, 8));
      float mn   = fmaxf(m8[j], mx);
      float corr = __expf(m8[j] - mn);
      float p0 = __expf(a - mn);
      float p1 = __expf(c - mn);
      float rs = p0 + p1;
      rs += __shfl_xor(rs, 1);
      rs += __shfl_xor(rs, 2);
      rs += __shfl_xor(rs, 4);
      rs += __shfl_xor(rs, 8);
      l8[j] = l8[j]*corr + rs;
      m8[j] = mn;
      o0[j] = o0[j]*corr;
      o1[j] = o1[j]*corr;
      int rm = j + 8*hi;
      sP[rm*32 + lr     ] = f32_bf16(p0);
      sP[rm*32 + 16 + lr] = f32_bf16(p1);
    }
    __syncthreads();
    const unsigned short* pr = &sP[lr*32 + hi*16];
    v16bf Ap = ld_bf16x16(pr, pr + 8);                   // P as 16q x 32k A fragment
    const unsigned short* v0p = vbase + (size_t)lr       * L + kb + hi*16;
    const unsigned short* v1p = vbase + (size_t)(16 + lr)* L + kb + hi*16;
    o0 = wmma_bf16(Ap, ld_bf16x16(v0p, v0p+8), o0);
    o1 = wmma_bf16(Ap, ld_bf16x16(v1p, v1p+8), o1);
    __syncthreads();
  }
  #pragma unroll
  for (int j=0;j<8;++j){
    int rm = j + 8*hi;
    float inv = 1.0f / l8[j];
    size_t o = ((size_t)(b*L + qb + rm))*256 + h*32;
    out[o + lr]      = o0[j]*inv;
    out[o + 16 + lr] = o1[j]*inv;
  }
}

// ---------------------------------------------------------------------------
// Small helpers
// ---------------------------------------------------------------------------
__global__ void cvt_bf16_kernel(const float* __restrict__ src, unsigned short* __restrict__ dst, int n){
  int i = blockIdx.x*256 + threadIdx.x;
  if (i < n) dst[i] = f32_bf16(src[i]);
}
__global__ void base_transpose_kernel(const float* __restrict__ src, unsigned short* __restrict__ dst){
  int i = blockIdx.x*256 + threadIdx.x;    // 256x256
  int in = i >> 8, o = i & 255;
  dst[o*256 + in] = f32_bf16(src[i]);
}
__global__ void wv_transpose_kernel(const unsigned short* __restrict__ wv, unsigned short* __restrict__ wvT){
  int i = blockIdx.x*256 + threadIdx.x;    // 2*2048*256
  int d = i & 255;
  int t = (i >> 8) & 2047;
  int b = i >> 19;
  int h = d >> 5, dd = d & 31;
  wvT[(((size_t)(b*8 + h)*32 + dd) * 2048) + t] = wv[i];
}
__global__ void gate_combine_kernel(const float* __restrict__ gate, float* __restrict__ o){
  int i = blockIdx.x*256 + threadIdx.x;
  o[i] = o[i] / (1.0f + __expf(-gate[i]));
}

// ---------------------------------------------------------------------------
extern "C" void kernel_launch(void* const* d_in, const int* in_sizes, int n_in,
                              void* d_out, int out_size, void* d_ws, size_t ws_size,
                              hipStream_t stream)
{
  (void)in_sizes; (void)n_in; (void)out_size; (void)ws_size;
  const float* q = (const float*)d_in[0];
  const float* k = (const float*)d_in[1];
  const float* v = (const float*)d_in[2];
  // layers: 0=p_q 1=p_k 2=p_v 3=p_g 4=p_o ; each: ln_s, ln_b, grid, denom, spline_w, base_w, base_b
  const float* P[5][7];
  for (int l=0;l<5;++l) for (int j=0;j<7;++j) P[l][j] = (const float*)d_in[3 + l*7 + j];

  char* w = (char*)d_ws; size_t off = 0;
  auto take = [&](size_t bytes)->void*{ void* p = w + off; off += (bytes + 255) & ~(size_t)255; return p; };
  const size_t NTOK = 4096;                          // B*L
  unsigned short* wq  = (unsigned short*)take(NTOK*256*2);
  unsigned short* wkb = (unsigned short*)take(NTOK*256*2);
  unsigned short* wvb = (unsigned short*)take(NTOK*256*2);
  unsigned short* wvT = (unsigned short*)take(NTOK*256*2);
  float* gate = (float*)take(NTOK*256*4);
  float* atto = (float*)take(NTOK*256*4);
  unsigned short* spl[5]; unsigned short* bas[5];
  for (int l=0;l<5;++l){ spl[l]=(unsigned short*)take((size_t)256*2048*2); bas[l]=(unsigned short*)take((size_t)256*256*2); }

  for (int l=0;l<5;++l){
    cvt_bf16_kernel<<<2048,256,0,stream>>>(P[l][4], spl[l], 256*2048);
    base_transpose_kernel<<<256,256,0,stream>>>(P[l][5], bas[l]);
  }
  const float qscale = 0.17677669529663687f;         // HD^-0.5, HD=32
  fastkan_kernel<<<256,256,0,stream>>>(q, P[0][0],P[0][1],P[0][2],P[0][3], spl[0], bas[0], P[0][6], qscale, wq,  nullptr);
  fastkan_kernel<<<256,256,0,stream>>>(k, P[1][0],P[1][1],P[1][2],P[1][3], spl[1], bas[1], P[1][6], 1.0f,   wkb, nullptr);
  fastkan_kernel<<<256,256,0,stream>>>(v, P[2][0],P[2][1],P[2][2],P[2][3], spl[2], bas[2], P[2][6], 1.0f,   wvb, nullptr);
  fastkan_kernel<<<256,256,0,stream>>>(q, P[3][0],P[3][1],P[3][2],P[3][3], spl[3], bas[3], P[3][6], 1.0f,   nullptr, gate);
  wv_transpose_kernel<<<4096,256,0,stream>>>(wvb, wvT);
  attn_kernel<<<dim3(128,8,2),32,0,stream>>>(wq, wkb, wvT, atto);
  gate_combine_kernel<<<4096,256,0,stream>>>(gate, atto);
  fastkan_kernel<<<256,256,0,stream>>>(atto, P[4][0],P[4][1],P[4][2],P[4][3], spl[4], bas[4], P[4][6], 1.0f, nullptr, (float*)d_out);
}
